// LiteMultiscaleAttention_2087354106475
// MI455X (gfx1250) — matrix operand order repaired and checked by reading, not compiled, not run on hardware
//
#include <hip/hip_runtime.h>
#include <hip/hip_bf16.h>
#include <stdint.h>

// ---------------------------------------------------------------------------
// LiteMultiscaleAttention on MI455X (gfx1250, wave32, WMMA + TDM).
// Big GEMMs (qkv 1x1 conv, proj 1x1 conv) use v_wmma_f32_16x16x32_bf16 with
// the B tile staged into LDS by the Tensor Data Mover (tensor_load_to_lds),
// double-buffered on TENSORcnt. A-operand (weights) loaded direct from L2.
// A-frag: row-major [M][K], per-lane 16 contiguous bf16 along K.
// B-frag: row-major [K][N] tile in LDS, per-lane column n=lane%16, 16 bf16 at
//         rows 16*(lane/16)+0..15 (LDS row pitch padded to 528B).
// ---------------------------------------------------------------------------

typedef __attribute__((ext_vector_type(16))) __bf16       v16bf;
typedef __attribute__((ext_vector_type(8)))  float        v8f;
typedef __attribute__((ext_vector_type(16))) uint16_t     us16;
typedef __attribute__((ext_vector_type(4)))  unsigned int v4u;
typedef __attribute__((ext_vector_type(8)))  int          v8i;
typedef __attribute__((ext_vector_type(4)))  int          v4i;

#define S_PIX 4096   // 64*64
#define C3    1536
#define CIN   512
#define NB    8

__device__ __forceinline__ uint16_t f2bf(float f) {
  uint32_t u = __builtin_bit_cast(uint32_t, f);
  u = (u + 0x7FFFu + ((u >> 16) & 1u)) >> 16;   // round-to-nearest-even
  return (uint16_t)u;
}
__device__ __forceinline__ float bf2f(uint16_t h) {
  uint32_t u = ((uint32_t)h) << 16;
  return __builtin_bit_cast(float, u);
}
__device__ __forceinline__ v16bf ldfrag(const uint16_t* p) {
  us16 u = *(const us16*)p;                      // 32B -> 2x b128 load
  return __builtin_bit_cast(v16bf, u);
}
__device__ __forceinline__ uint32_t lds_off_of(const void* p) {
  // Flat pointers to LDS carry the offset in addr[31:0] (ISA 10.2 aperture).
  return (uint32_t)(uintptr_t)p;
}

// TDM: DMA a [32 K][256 N] bf16 tile (row stride 4096 elems) into LDS with a
// 16B pad inserted after every 512B row (LDS pitch 528B = 264 uint16).
__device__ __forceinline__ void tdm_load_tile(uint32_t lds_off,
                                              const uint16_t* gsrc,
                                              int tensor_k) {
  uint64_t ga = (uint64_t)(uintptr_t)gsrc;
  v4u g0;
  g0[0] = 1u;                                             // count=1, user mode
  g0[1] = lds_off;                                        // lds_addr
  g0[2] = (uint32_t)ga;                                   // global_addr[31:0]
  g0[3] = (uint32_t)((ga >> 32) & 0x1FFFFFFu) | (2u << 30); // [56:32], type=2
  v8i g1 = {};
  // data_size=2B (code 1), pad_enable, pad_interval=512B (code 6),
  // pad_amount=4 DWORDs (code 3), workgroup_mask=0 (not in a cluster).
  g1[0] = (int)((1u << 16) | (1u << 20) | (6u << 22) | (3u << 25));
  g1[1] = (int)(4096u << 16);                             // tensor_dim0 = 4096
  g1[2] = (int)(((uint32_t)tensor_k & 0xFFFFu) << 16);    // tensor_dim1 = K
  g1[3] = (int)(256u << 16);                              // tile_dim0 = 256
  g1[4] = 32;                                             // tile_dim1 = 32
  g1[5] = 4096;                                           // tensor_dim0_stride
  v4i gz = {};
#if __clang_major__ >= 23
  v8i gz8 = {};
  __builtin_amdgcn_tensor_load_to_lds(g0, g1, gz, gz, gz8, 0);
#else
  __builtin_amdgcn_tensor_load_to_lds(g0, g1, gz, gz, 0);
#endif
}

// ------------------------------ converts -----------------------------------
__global__ void cvt_f32_to_bf16(const float* __restrict__ src,
                                uint16_t* __restrict__ dst, int n) {
  int i = blockIdx.x * blockDim.x + threadIdx.x;
  int stride = gridDim.x * blockDim.x;
  for (; i < n; i += stride) dst[i] = f2bf(src[i]);
}

__global__ void bn_prep(const float* __restrict__ g, const float* __restrict__ b,
                        const float* __restrict__ m, const float* __restrict__ v,
                        float* __restrict__ inv, float* __restrict__ add) {
  int i = blockIdx.x * blockDim.x + threadIdx.x;
  if (i < 512) {
    float iv = g[i] * rsqrtf(v[i] + 1e-5f);
    inv[i] = iv;
    add[i] = b[i] - m[i] * iv;
  }
}

// ---------------- WMMA GEMM with TDM-staged B operand ----------------------
// D[M x 4096] = A[M x K] * B[K x 4096] per batch; block tile 64M x 256N,
// 4 waves each own a 64M x 64N sub-tile (4x4 wmma accumulators).
// BN=false: bf16 store.  BN=true: fused BatchNorm affine, fp32 store.
template <int K, bool BN>
__launch_bounds__(128)
__global__ void gemm_bf16_wmma_tdm(const uint16_t* __restrict__ A,
                                   const uint16_t* __restrict__ Bmat,
                                   uint16_t* __restrict__ Cbf,
                                   const float* __restrict__ bninv,
                                   const float* __restrict__ bnadd,
                                   float* __restrict__ Yf) {
  __shared__ __align__(32) uint16_t tileB[2][32][264];  // 528B row pitch
  const int lane = threadIdx.x & 31, wave = threadIdx.x >> 5;
  const int lr = lane & 15, lh = lane >> 4;
  const int m0   = blockIdx.x * 64;
  const int nblk = blockIdx.y * 256;
  const int b    = blockIdx.z;
  const uint16_t* Bb = Bmat + (size_t)b * K * S_PIX;

  v8f acc[4][4] = {};
  if (wave == 0) tdm_load_tile(lds_off_of(&tileB[0][0][0]), Bb + nblk, K);

  int cur = 0;
  for (int k0 = 0; k0 < K; k0 += 32, cur ^= 1) {
    if (wave == 0) {
      if (k0 + 32 < K) {
        tdm_load_tile(lds_off_of(&tileB[cur ^ 1][0][0]),
                      Bb + (size_t)(k0 + 32) * S_PIX + nblk, K);
        __builtin_amdgcn_s_wait_tensorcnt(1);  // oldest (current buf) landed
      } else {
        __builtin_amdgcn_s_wait_tensorcnt(0);
      }
    }
    __syncthreads();   // publish tileB[cur] to all waves

    v16bf a[4], bb[4];
#pragma unroll
    for (int mi = 0; mi < 4; ++mi)
      a[mi] = ldfrag(A + (size_t)(m0 + mi * 16 + lr) * K + k0 + lh * 16);
#pragma unroll
    for (int ni = 0; ni < 4; ++ni) {
      us16 u;
#pragma unroll
      for (int s = 0; s < 16; ++s)
        u[s] = tileB[cur][lh * 16 + s][wave * 64 + ni * 16 + lr];
      bb[ni] = __builtin_bit_cast(v16bf, u);
    }
#pragma unroll
    for (int mi = 0; mi < 4; ++mi)
#pragma unroll
      for (int ni = 0; ni < 4; ++ni)
        acc[mi][ni] = __builtin_amdgcn_wmma_f32_16x16x32_bf16(
            false, a[mi], false, bb[ni], (short)0, acc[mi][ni], false, false);

    __syncthreads();   // all waves done reading before buffer is re-filled
  }

  const int n0 = nblk + wave * 64;
#pragma unroll
  for (int mi = 0; mi < 4; ++mi)
#pragma unroll
    for (int ni = 0; ni < 4; ++ni)
#pragma unroll
      for (int r = 0; r < 8; ++r) {
        int m = m0 + mi * 16 + r + 8 * lh;   // C layout: M = r + 8*(lane/16)
        int n = n0 + ni * 16 + lr;           //           N = lane%16
        if (BN) {
          Yf[((size_t)b * 512 + m) * S_PIX + n] =
              acc[mi][ni][r] * bninv[m] + bnadd[m];
        } else {
          Cbf[((size_t)b * C3 + m) * S_PIX + n] = f2bf(acc[mi][ni][r]);
        }
      }
}

// --------------------- depthwise 5x5, SAME, per-channel --------------------
__global__ void dwconv5(const uint16_t* __restrict__ Q, const float* __restrict__ W,
                        uint16_t* __restrict__ O) {
  const int bc = blockIdx.x;          // b*1536 + c
  const int c  = bc % C3;
  const uint16_t* src = Q + (size_t)bc * S_PIX;
  uint16_t* dst = O + (size_t)bc * S_PIX;
  const float* w = W + c * 25;        // uniform per block -> scalar loads
  for (int p = threadIdx.x; p < S_PIX; p += blockDim.x) {
    int x = p & 63, y = p >> 6;
    float acc = 0.f;
#pragma unroll
    for (int ky = 0; ky < 5; ++ky) {
      int yy = y + ky - 2;
      if (yy < 0 || yy > 63) continue;
#pragma unroll
      for (int kx = 0; kx < 5; ++kx) {
        int xx = x + kx - 2;
        if (xx < 0 || xx > 63) continue;
        acc += w[ky * 5 + kx] * bf2f(src[yy * 64 + xx]);
      }
    }
    dst[p] = f2bf(acc);
  }
}

// --------------- grouped 1x1 conv: 48 groups of 32 -> 32 -------------------
__global__ void pwconv(const uint16_t* __restrict__ D, const float* __restrict__ W,
                       uint16_t* __restrict__ Agg) {
  const int s = blockIdx.x * 256 + threadIdx.x;    // grid.x = 16
  const int g = blockIdx.y, b = blockIdx.z;
  const uint16_t* base = D + ((size_t)b * C3 + g * 32) * S_PIX + s;
  float xv[32];
#pragma unroll
  for (int i = 0; i < 32; ++i) xv[i] = bf2f(base[(size_t)i * S_PIX]);
  uint16_t* obase = Agg + ((size_t)b * C3 + g * 32) * S_PIX + s;
  for (int o = 0; o < 32; ++o) {
    const float* w = W + (g * 32 + o) * 32;        // uniform -> scalar loads
    float acc = 0.f;
#pragma unroll
    for (int i = 0; i < 32; ++i) acc += w[i] * xv[i];
    obase[(size_t)o * S_PIX] = f2bf(acc);
  }
}

// ------------------------ fused linear attention ---------------------------
// One block per (head, batch). cat channels: head h<32 -> qkv buf, else agg.
// Within 48 rows: q=0..15, k=16..31, v=32..47.  vk = [v;1] @ relu(k)^T (17x16),
// out = vk @ relu(q), normalized by the ones-row, bf16 out in [C][S] layout.
__launch_bounds__(256)
__global__ void lin_attention(const uint16_t* __restrict__ QKV,
                              const uint16_t* __restrict__ AGG,
                              uint16_t* __restrict__ OUT) {
  __shared__ float smem[8192];
  const int h = blockIdx.x, b = blockIdx.y;
  const int t = threadIdx.x;
  const int e = t & 15, sub = t >> 4;
  const uint16_t* base = (h < 32)
      ? QKV + ((size_t)b * C3 + h * 48) * S_PIX
      : AGG + ((size_t)b * C3 + (h - 32) * 48) * S_PIX;
  const uint16_t* qb = base;
  const uint16_t* kb = base + (size_t)16 * S_PIX;
  const uint16_t* vb = base + (size_t)32 * S_PIX;

  float acc[17];
#pragma unroll
  for (int d = 0; d < 17; ++d) acc[d] = 0.f;

  float* ks = smem;          // [256][16]
  float* vs = smem + 4096;   // [256][16]

  // Phase 1: vk partials. Thread (e,sub) accumulates vk[:,e] over its s-subset.
  for (int s0 = 0; s0 < S_PIX; s0 += 256) {
#pragma unroll
    for (int r = 0; r < 16; ++r) {
      float kv = bf2f(kb[(size_t)r * S_PIX + s0 + t]);
      ks[t * 16 + r] = fmaxf(kv, 0.f);
      vs[t * 16 + r] = bf2f(vb[(size_t)r * S_PIX + s0 + t]);
    }
    __syncthreads();
#pragma unroll
    for (int j = 0; j < 16; ++j) {
      int sl = sub * 16 + j;
      float kv = ks[sl * 16 + e];
#pragma unroll
      for (int d = 0; d < 16; ++d) acc[d] += vs[sl * 16 + d] * kv;
      acc[16] += kv;   // ones row of padded v
    }
    __syncthreads();
  }

  // Reduce 16 sub-partials per (d,e): red[(d*16+e)*16 + sub], max idx 4351.
#pragma unroll
  for (int d = 0; d < 17; ++d) smem[(d * 16 + e) * 16 + sub] = acc[d];
  __syncthreads();
  float* vkf = smem + 4352;  // final vk[17][16], disjoint from red region
  for (int p = t; p < 272; p += 256) {
    float sum = 0.f;
#pragma unroll
    for (int s = 0; s < 16; ++s) sum += smem[p * 16 + s];
    vkf[p] = sum;
  }
  __syncthreads();

  // Phase 2: out[d,s] = (vk[d,:]·relu(q[:,s])) / (vk[16,:]·relu(q[:,s]) + eps)
  for (int i = 0; i < 16; ++i) {
    int s = i * 256 + t;
    float qv[16];
#pragma unroll
    for (int r = 0; r < 16; ++r)
      qv[r] = fmaxf(bf2f(qb[(size_t)r * S_PIX + s]), 0.f);
    float num[17];
#pragma unroll
    for (int d = 0; d < 17; ++d) {
      float a = 0.f;
#pragma unroll
      for (int r = 0; r < 16; ++r) a += vkf[d * 16 + r] * qv[r];
      num[d] = a;
    }
    float rden = 1.0f / (num[16] + 1e-15f);
#pragma unroll
    for (int d = 0; d < 16; ++d)
      OUT[((size_t)b * 1024 + h * 16 + d) * S_PIX + s] = f2bf(num[d] * rden);
  }
}

// ------------------------------- launch ------------------------------------
extern "C" void kernel_launch(void* const* d_in, const int* in_sizes, int n_in,
                              void* d_out, int out_size, void* d_ws, size_t ws_size,
                              hipStream_t stream) {
  (void)in_sizes; (void)n_in; (void)out_size; (void)ws_size;
  const float* x      = (const float*)d_in[0];
  const float* w_qkv  = (const float*)d_in[1];
  const float* w_dw   = (const float*)d_in[2];
  const float* w_pw   = (const float*)d_in[3];
  const float* w_proj = (const float*)d_in[4];
  const float* bn_g   = (const float*)d_in[5];
  const float* bn_b   = (const float*)d_in[6];
  const float* bn_m   = (const float*)d_in[7];
  const float* bn_v   = (const float*)d_in[8];

  char* ws = (char*)d_ws;
  size_t off = 0;
  auto take = [&](size_t bytes) -> char* {
    char* p = ws + off;
    off = (off + bytes + 255) & ~(size_t)255;
    return p;
  };
  uint16_t* wqkv_bf  = (uint16_t*)take(2ull * 1536 * 512);
  uint16_t* wproj_bf = (uint16_t*)take(2ull * 512 * 1024);
  float*    bninv    = (float*)take(512 * 4);
  float*    bnadd    = (float*)take(512 * 4);
  uint16_t* x_bf     = (uint16_t*)take(2ull * NB * CIN * S_PIX);
  uint16_t* qkv_bf   = (uint16_t*)take(2ull * NB * C3 * S_PIX);
  uint16_t* dw_bf    = (uint16_t*)take(2ull * NB * C3 * S_PIX);
  uint16_t* agg_bf   = (uint16_t*)take(2ull * NB * C3 * S_PIX);
  uint16_t* out_bf   = dw_bf;   // dw is dead after pwconv; reuse for attn out

  cvt_f32_to_bf16<<<2048, 256, 0, stream>>>(x, x_bf, NB * CIN * S_PIX);
  cvt_f32_to_bf16<<<512, 256, 0, stream>>>(w_qkv, wqkv_bf, 1536 * 512);
  cvt_f32_to_bf16<<<512, 256, 0, stream>>>(w_proj, wproj_bf, 512 * 1024);
  bn_prep<<<2, 256, 0, stream>>>(bn_g, bn_b, bn_m, bn_v, bninv, bnadd);

  gemm_bf16_wmma_tdm<512, false><<<dim3(24, 16, 8), 128, 0, stream>>>(
      wqkv_bf, x_bf, qkv_bf, nullptr, nullptr, nullptr);
  dwconv5<<<NB * C3, 256, 0, stream>>>(qkv_bf, w_dw, dw_bf);
  pwconv<<<dim3(16, 48, 8), 256, 0, stream>>>(dw_bf, w_pw, agg_bf);
  lin_attention<<<dim3(64, 8), 256, 0, stream>>>(qkv_bf, agg_bf, out_bf);
  gemm_bf16_wmma_tdm<1024, true><<<dim3(8, 16, 8), 128, 0, stream>>>(
      wproj_bf, out_bf, nullptr, bninv, bnadd, (float*)d_out);
}
